// Conv_KNRM_20564303413359
// MI455X (gfx1250) — matrix operand order repaired and checked by reading, not compiled, not run on hardware
//
#include <hip/hip_runtime.h>
#include <hip/hip_bf16.h>

typedef __attribute__((ext_vector_type(16))) _Float16 v16h;
typedef __attribute__((ext_vector_type(8)))  _Float16 v8h;
typedef __attribute__((ext_vector_type(8)))  float    v8f;

#define B_   128
#define LQ_  64
#define LD_  512
#define E_   300
#define EP_  320   // E padded to multiple of 32 (K granularity of WMMA f16)
#define O_   128
#define NK_  11
#define CW_  8     // waves per block in cosine kernel (8 x 64 cols = 512)

// RBF kernel constants: mu = 1.0,0.9,0.7,...,-0.9 ; 1/(2*sigma^2): 5e5 then 50
__constant__ float MU_C[NK_] = {1.f,0.9f,0.7f,0.5f,0.3f,0.1f,-0.1f,-0.3f,-0.5f,-0.7f,-0.9f};
__constant__ float IT_C[NK_] = {500000.f,50.f,50.f,50.f,50.f,50.f,50.f,50.f,50.f,50.f,50.f};

// ---------------- embedding gather: tokens -> (B, L+2, EP) f16, zero padded ----
__global__ void embed_gather(const int* __restrict__ tok, const float* __restrict__ emb,
                             _Float16* __restrict__ dst, int L) {
  int row = blockIdx.x;                 // b*(L+2)+t
  int b = row / (L + 2), t = row % (L + 2);
  int e = threadIdx.x;                  // 0..EP_-1
  _Float16 v = (_Float16)0.0f;
  if (t < L && e < E_) {
    int tk = tok[b * L + t];
    v = (_Float16)emb[(size_t)tk * E_ + e];
  }
  dst[(size_t)row * EP_ + e] = v;
}

// ---------------- weight repack: (O,E,h) f32 -> (O, h*EP) f16 -------------------
__global__ void pack_weights(const float* __restrict__ w, _Float16* __restrict__ dst, int h) {
  int idx = blockIdx.x * blockDim.x + threadIdx.x;
  int KH = h * EP_;
  if (idx >= O_ * KH) return;
  int o = idx / KH, r = idx % KH;
  int k = r / EP_, e = r % EP_;
  float v = (e < E_) ? w[(size_t)o * E_ * h + (size_t)e * h + k] : 0.0f;
  dst[idx] = (_Float16)v;
}

// ---------------- conv-as-GEMM (im2col view) + bias + relu -> f16 ---------------
// block: (32, 2) waves; each wave computes a 16x64 strip (4 N-tiles) of (L x O).
// One A-fragment load feeds 4 WMMAs.
__global__ void conv_gemm(const _Float16* __restrict__ x,   // (B, L+2, EP)
                          const _Float16* __restrict__ w2,  // (O, KH)
                          const float* __restrict__ bias,
                          _Float16* __restrict__ y,         // (B, L, O)
                          int L, int KH) {
  int lane = threadIdx.x;
  int n0 = threadIdx.y * 64;            // 16x64 strip per wave
  int m0 = blockIdx.x * 16;
  int b  = blockIdx.y;
  int hl = lane >> 4, lr = lane & 15;
  const _Float16* abase = x + ((size_t)b * (L + 2) + (m0 + lr)) * EP_;
  const _Float16* bb0 = w2 + (size_t)(n0 + lr)      * KH + hl * 16;
  const _Float16* bb1 = w2 + (size_t)(n0 + 16 + lr) * KH + hl * 16;
  const _Float16* bb2 = w2 + (size_t)(n0 + 32 + lr) * KH + hl * 16;
  const _Float16* bb3 = w2 + (size_t)(n0 + 48 + lr) * KH + hl * 16;
  int sa = hl * 16;
  v8f c0 = {}, c1 = {}, c2 = {}, c3 = {};
  for (int kt = 0; kt < KH; kt += 32) {
    v8h alo = *(const v8h*)(abase + kt + sa);
    v8h ahi = *(const v8h*)(abase + kt + 8 + sa);
    v16h a = __builtin_shufflevector(alo, ahi, 0,1,2,3,4,5,6,7,8,9,10,11,12,13,14,15);
    v16h bm0 = *(const v16h*)(bb0 + kt);
    v16h bm1 = *(const v16h*)(bb1 + kt);
    v16h bm2 = *(const v16h*)(bb2 + kt);
    v16h bm3 = *(const v16h*)(bb3 + kt);
    c0 = __builtin_amdgcn_wmma_f32_16x16x32_f16(false, a, false, bm0, (short)0, c0, false, false);
    c1 = __builtin_amdgcn_wmma_f32_16x16x32_f16(false, a, false, bm1, (short)0, c1, false, false);
    c2 = __builtin_amdgcn_wmma_f32_16x16x32_f16(false, a, false, bm2, (short)0, c2, false, false);
    c3 = __builtin_amdgcn_wmma_f32_16x16x32_f16(false, a, false, bm3, (short)0, c3, false, false);
  }
  _Float16* ybase = y + ((size_t)b * L + m0 + hl * 8) * O_;
  v8f* cs[4] = {&c0, &c1, &c2, &c3};
#pragma unroll
  for (int t = 0; t < 4; ++t) {
    int col = n0 + t * 16 + lr;
    float bs = bias[col];
    v8f cv = *cs[t];
#pragma unroll
    for (int r = 0; r < 8; ++r) {
      float v = cv[r] + bs;
      v = v > 0.f ? v : 0.f;
      ybase[(size_t)r * O_ + col] = (_Float16)v;
    }
  }
}

// ---------------- per-row inverse L2 norm: rn = 1/(||row||+1e-13) ---------------
__global__ void row_rnorm(const _Float16* __restrict__ h, float* __restrict__ rn) {
  int row = blockIdx.x;
  int lane = threadIdx.x;
  const _Float16* p = h + (size_t)row * O_;
  float s = 0.f;
  for (int i = lane; i < O_; i += 32) { float v = (float)p[i]; s += v * v; }
#pragma unroll
  for (int d = 16; d >= 1; d >>= 1) s += __shfl_xor(s, d, 32);
  if (lane == 0) rn[row] = 1.f / (sqrtf(s) + 1e-13f);
}

// ---------------- cosine GEMM + fused RBF pooling -------------------------------
// grid: (1, LQ/16, 9*B), block: (32, CW_). Each wave: 16x64 strip; the 8 waves
// together cover all 512 doc positions, so the block reduces its kernel sums in
// LDS and stores qk with plain stores (no atomics, no zero pass).
__global__ void cos_rbf(const _Float16* __restrict__ hq, const _Float16* __restrict__ hd,
                        const float* __restrict__ rnq, const float* __restrict__ rnd,
                        const int* __restrict__ tq, const int* __restrict__ td,
                        float* __restrict__ qk) {
  __shared__ float smv[CW_][256];
  __shared__ float smm[CW_][256];
  __shared__ float qacc[CW_][16][NK_];
  int lane = threadIdx.x, w = threadIdx.y;
  int wn0 = w * 64;                     // this wave's 64-column slab
  int m0 = blockIdx.y * 16;
  int pb = blockIdx.z;
  int b = pb % B_, pair = pb / B_;
  int qg = pair / 3, dg = pair % 3;
  int hl = lane >> 4, lr = lane & 15;
  const _Float16* ab = hq + (((size_t)(qg * B_ + b)) * LQ_ + m0 + lr) * O_;
  const _Float16* hdb = hd + ((size_t)(dg * B_ + b)) * LD_ * O_;
  const _Float16* bb0 = hdb + (size_t)(wn0 + lr)      * O_ + hl * 16;
  const _Float16* bb1 = hdb + (size_t)(wn0 + 16 + lr) * O_ + hl * 16;
  const _Float16* bb2 = hdb + (size_t)(wn0 + 32 + lr) * O_ + hl * 16;
  const _Float16* bb3 = hdb + (size_t)(wn0 + 48 + lr) * O_ + hl * 16;
  int sa = hl * 16;
  v8f c0 = {}, c1 = {}, c2 = {}, c3 = {};
#pragma unroll
  for (int kt = 0; kt < O_; kt += 32) {
    v8h alo = *(const v8h*)(ab + kt + sa);
    v8h ahi = *(const v8h*)(ab + kt + 8 + sa);
    v16h a = __builtin_shufflevector(alo, ahi, 0,1,2,3,4,5,6,7,8,9,10,11,12,13,14,15);
    v16h bm0 = *(const v16h*)(bb0 + kt);
    v16h bm1 = *(const v16h*)(bb1 + kt);
    v16h bm2 = *(const v16h*)(bb2 + kt);
    v16h bm3 = *(const v16h*)(bb3 + kt);
    c0 = __builtin_amdgcn_wmma_f32_16x16x32_f16(false, a, false, bm0, (short)0, c0, false, false);
    c1 = __builtin_amdgcn_wmma_f32_16x16x32_f16(false, a, false, bm1, (short)0, c1, false, false);
    c2 = __builtin_amdgcn_wmma_f32_16x16x32_f16(false, a, false, bm2, (short)0, c2, false, false);
    c3 = __builtin_amdgcn_wmma_f32_16x16x32_f16(false, a, false, bm3, (short)0, c3, false, false);
  }
  // per-lane row norms/masks for the 16 q rows this wave touches
  float acc[NK_];
#pragma unroll
  for (int k = 0; k < NK_; ++k) acc[k] = 0.f;
  v8f cs[4] = {c0, c1, c2, c3};
#pragma unroll
  for (int t = 0; t < 4; ++t) {
    int col = wn0 + t * 16 + lr;
    float rd = rnd[(size_t)(dg * B_ + b) * LD_ + col];
    float dm = (td[b * LD_ + col] > 0) ? 1.f : 0.f;
#pragma unroll
    for (int r = 0; r < 8; ++r) {
      int row = r + hl * 8;
      int q = m0 + row;
      float rq = rnq[(size_t)(qg * B_ + b) * LQ_ + q];
      float qm = (tq[b * LQ_ + q] > 0) ? 1.f : 0.f;
      float msk = qm * dm;
      smv[w][row * 16 + lr] = cs[t][r] * rq * rd * msk;
      smm[w][row * 16 + lr] = msk;
    }
    __syncthreads();
    // lanes 0-15: rows 0-15, cols 0-7 ; lanes 16-31: rows 0-15, cols 8-15
#pragma unroll
    for (int cc = 0; cc < 8; ++cc) {
      int cidx = hl * 8 + cc;
      float m = smv[w][lr * 16 + cidx];
      float mk = smm[w][lr * 16 + cidx];
#pragma unroll
      for (int k = 0; k < NK_; ++k) {
        float d = m - MU_C[k];
        acc[k] += __expf(-d * d * IT_C[k]) * mk;
      }
    }
    __syncthreads();
  }
  // combine the two column-halves, park per-wave row sums in LDS
#pragma unroll
  for (int k = 0; k < NK_; ++k) {
    float v = acc[k] + __shfl_down(acc[k], 16, 32);
    if (hl == 0) qacc[w][lr][k] = v;
  }
  __syncthreads();
  // cross-wave reduce and store qk (each (pair,b,row) owned by exactly one block)
  int tid = w * 32 + lane;
  if (tid < 16 * NK_) {
    int row = tid / NK_, k = tid % NK_;
    float s = 0.f;
#pragma unroll
    for (int ww = 0; ww < CW_; ++ww) s += qacc[ww][row][k];
    qk[((size_t)(pair * B_ + b) * LQ_ + m0 + row) * NK_ + k] = s;
  }
}

// ---------------- log pooling + FC ---------------------------------------------
__global__ void finalize_fc(const float* __restrict__ qk, const float* __restrict__ fcw,
                            const float* __restrict__ fcb, float* __restrict__ out) {
  __shared__ float red[128];
  int b = blockIdx.x, f = threadIdx.x;
  float v = 0.f;
  if (f < 99) {
    int pair = f / NK_, k = f % NK_;
    const float* p = qk + ((size_t)(pair * B_ + b) * LQ_) * NK_ + k;
    float s = 0.f;
    for (int q = 0; q < LQ_; ++q) s += logf(fmaxf(p[q * NK_], 1e-10f)) * 0.01f;
    v = s * fcw[f];
  }
  red[f] = v;
  __syncthreads();
  for (int d = 64; d >= 1; d >>= 1) {
    if (f < d) red[f] += red[f + d];
    __syncthreads();
  }
  if (f == 0) out[b] = red[0] + fcb[0];
}

extern "C" void kernel_launch(void* const* d_in, const int* in_sizes, int n_in,
                              void* d_out, int out_size, void* d_ws, size_t ws_size,
                              hipStream_t stream) {
  const int*   tq  = (const int*)d_in[0];
  const int*   td  = (const int*)d_in[1];
  const float* emb = (const float*)d_in[2];
  const float* cw[3] = {(const float*)d_in[3], (const float*)d_in[5], (const float*)d_in[7]};
  const float* cb[3] = {(const float*)d_in[4], (const float*)d_in[6], (const float*)d_in[8]};
  const float* fcw = (const float*)d_in[9];
  const float* fcb = (const float*)d_in[10];
  float* out = (float*)d_out;

  char* ws = (char*)d_ws;
  size_t off = 0;
  auto take = [&](size_t bytes) { char* p = ws + off; off = (off + bytes + 255) & ~(size_t)255; return p; };

  _Float16* qe  = (_Float16*)take((size_t)B_ * (LQ_ + 2) * EP_ * 2);
  _Float16* de  = (_Float16*)take((size_t)B_ * (LD_ + 2) * EP_ * 2);
  _Float16* w2g[3];
  for (int g = 0; g < 3; ++g) w2g[g] = (_Float16*)take((size_t)O_ * (g + 1) * EP_ * 2);
  _Float16* hq  = (_Float16*)take((size_t)3 * B_ * LQ_ * O_ * 2);
  _Float16* hd  = (_Float16*)take((size_t)3 * B_ * LD_ * O_ * 2);
  float*    rnq = (float*)take((size_t)3 * B_ * LQ_ * 4);
  float*    rnd = (float*)take((size_t)3 * B_ * LD_ * 4);
  float*    qk  = (float*)take((size_t)9 * B_ * LQ_ * NK_ * 4);

  // 1) gather embeddings to padded f16
  embed_gather<<<B_ * (LQ_ + 2), EP_, 0, stream>>>(tq, emb, qe, LQ_);
  embed_gather<<<B_ * (LD_ + 2), EP_, 0, stream>>>(td, emb, de, LD_);

  // 2) repack conv weights to (O, h*EP) f16
  for (int g = 0; g < 3; ++g) {
    int h = g + 1, tot = O_ * h * EP_;
    pack_weights<<<(tot + 255) / 256, 256, 0, stream>>>(cw[g], w2g[g], h);
  }

  // 3) conv GEMMs (WMMA), relu, store f16 activations
  for (int g = 0; g < 3; ++g) {
    int h = g + 1, KH = h * EP_;
    conv_gemm<<<dim3(LQ_ / 16, B_), dim3(32, 2), 0, stream>>>(
        qe, w2g[g], cb[g], hq + (size_t)g * B_ * LQ_ * O_, LQ_, KH);
    conv_gemm<<<dim3(LD_ / 16, B_), dim3(32, 2), 0, stream>>>(
        de, w2g[g], cb[g], hd + (size_t)g * B_ * LD_ * O_, LD_, KH);
  }

  // 4) inverse row norms for cosine
  row_rnorm<<<3 * B_ * LQ_, 32, 0, stream>>>(hq, rnq);
  row_rnorm<<<3 * B_ * LD_, 32, 0, stream>>>(hd, rnd);

  // 5) cosine GEMM + fused RBF pooling (block owns full doc dimension -> no atomics)
  cos_rbf<<<dim3(1, LQ_ / 16, 9 * B_), dim3(32, CW_), 0, stream>>>(
      hq, hd, rnq, rnd, tq, td, qk);

  // 6) log pooling + FC
  finalize_fc<<<B_, 128, 0, stream>>>(qk, fcw, fcb, out);

  (void)in_sizes; (void)n_in; (void)out_size; (void)ws_size;
}
